// VQCrossAttenSeg_62302795596232
// MI455X (gfx1250) — matrix operand order repaired and checked by reading, not compiled, not run on hardware
//
#include <hip/hip_runtime.h>
#include <hip/hip_bf16.h>
#include <math.h>

// ---------------------------------------------------------------------------
// VQ cross-attention segment for MI455X (gfx1250, wave32, WMMA f16)
//   q      = x @ Wq^T                  [65536, 256]
//   k      = env @ Wk^T                [512, 256]   (f16, precomputed)
//   v      = env @ Wv^T                [512]        (f32, precomputed)
//   logits = (q @ k^T) * 256^-0.5 + gumbel          [65536, 512]
//   out    = (sigmoid(v[argmax]), one_hot(argmax))  (forward value of ST-gumbel)
//
// Roofline: ~335 MB of single-use HBM traffic @ 23.3 TB/s => ~14.4 us floor;
// 26 GFLOP via f16 WMMA is negligible at that point. Streaming tensors
// (x, gumbel, env_selects) use NT cache policy so the L2 keeps the reused
// Wq_h/k_h matrices (~1.5 GB of refetch demand) resident.
// ---------------------------------------------------------------------------

typedef _Float16 v16h __attribute__((ext_vector_type(16)));
typedef _Float16 v8h  __attribute__((ext_vector_type(8)));
typedef float    v8f  __attribute__((ext_vector_type(8)));
typedef float    v4f  __attribute__((ext_vector_type(4)));

#define BL      65536   // B*L
#define CDIM    256
#define QKDIM   256
#define NDIM    512
#define DDIM    256
#define SCALING 0.0625f // 256^-0.5

// ---------------- prep: Wq f32 -> f16 ----------------
__global__ void wq_to_f16(const float* __restrict__ Wq, _Float16* __restrict__ Wq_h) {
    int gtid = blockIdx.x * 256 + threadIdx.x;       // 16384 threads
#pragma unroll
    for (int j = 0; j < 4; ++j) {
        int i = gtid + j * 16384;                    // 65536 elements total
        Wq_h[i] = (_Float16)Wq[i];
    }
}

// ---------------- prep: k = env @ Wk^T (f16), v = env @ Wv^T (f32) ----------
__global__ void prep_kv(const float* __restrict__ env, const float* __restrict__ Wk,
                        const float* __restrict__ Wv, _Float16* __restrict__ k_h,
                        float* __restrict__ vout) {
    __shared__ float row[DDIM];
    __shared__ float part[256];
    const int n = blockIdx.x;          // 0..511
    const int t = threadIdx.x;         // 0..255
    row[t] = env[(size_t)n * DDIM + t];
    __syncthreads();
    const float* wk = Wk + (size_t)t * DDIM;   // Wk[t, :]
    float s = 0.0f;
#pragma unroll 8
    for (int d = 0; d < DDIM; ++d) s = fmaf(row[d], wk[d], s);
    k_h[(size_t)n * DDIM + t] = (_Float16)s;
    part[t] = row[t] * Wv[t];
    __syncthreads();
    for (int off = 128; off > 0; off >>= 1) {
        if (t < off) part[t] += part[t + off];
        __syncthreads();
    }
    if (t == 0) vout[n] = part[0];
}

// ---------------- main fused kernel ----------------
// 256 threads = 8 waves; wave w owns rows [ (blockIdx*8+w)*16 , +16 ).
__global__ void __launch_bounds__(256)
vq_cross_atten(const float* __restrict__ x, const _Float16* __restrict__ Wq_h,
               const _Float16* __restrict__ k_h, const float* __restrict__ vvec,
               const float* __restrict__ gumbel, float* __restrict__ env_att,
               float* __restrict__ env_sel) {
    __shared__ _Float16 q_lds[8 * 16 * QKDIM];       // 64 KB: per-wave q tiles

    const int wave = threadIdx.x >> 5;
    const int lane = threadIdx.x & 31;
    const int hg   = lane >> 4;          // half-group: 0 = lanes 0-15, 1 = lanes 16-31
    const int l16  = lane & 15;
    const int row0 = (blockIdx.x * 8 + wave) * 16;   // tile base row in [0, BL)

    // ---- Phase 1: load x A-fragments (contiguous-k 16-bit A layout, 16x32 per chunk) ----
    // lanes 0-15: M=lane, halves 0-15 -> K=kc*32+0..15
    // lanes16-31: M=lane-16, halves 0-15 -> K=kc*32+16..31
    // x is read exactly once kernel-wide -> non-temporal loads.
    v16h af[8];
    {
        const float* xr = x + (size_t)(row0 + l16) * CDIM;
#pragma unroll
        for (int kc = 0; kc < 8; ++kc) {
            const int base = kc * 32 + hg * 16;
            const v4f a0 = __builtin_nontemporal_load((const v4f*)(xr + base + 0));
            const v4f a1 = __builtin_nontemporal_load((const v4f*)(xr + base + 4));
            const v4f b0 = __builtin_nontemporal_load((const v4f*)(xr + base + 8));
            const v4f b1 = __builtin_nontemporal_load((const v4f*)(xr + base + 12));
            v16h a;
#pragma unroll
            for (int i = 0; i < 4; ++i) {
                a[i]      = (_Float16)a0[i];
                a[4 + i]  = (_Float16)a1[i];
                a[8 + i]  = (_Float16)b0[i];
                a[12 + i] = (_Float16)b1[i];
            }
            af[kc] = a;
        }
    }

    // ---- Phase 1b: q[16,256] = x @ Wq^T via WMMA, spill to LDS row-major ----
    // B layout per lane: N = l16, K = hg*16 + 0..15 contiguous -> 32B load.
    // B[k,n] = Wq[qb*16+n, kc*32+k]  (Wq is [out,in] row-major; reused -> RT policy).
    {
        _Float16* qs = q_lds + wave * 16 * QKDIM;
#pragma unroll
        for (int qb = 0; qb < 16; ++qb) {
            v8f acc = {};
            const _Float16* wqb = Wq_h + (size_t)(qb * 16 + l16) * CDIM + hg * 16;
#pragma unroll
            for (int kc = 0; kc < 8; ++kc) {
                const v16h b = *(const v16h*)(wqb + kc * 32);
                acc = __builtin_amdgcn_wmma_f32_16x16x32_f16(
                    false, af[kc], false, b, (short)0, acc, false, false);
            }
            // D layout: VGPR r -> row M = r + hg*8, column N = l16 of block qb
#pragma unroll
            for (int r = 0; r < 8; ++r)
                qs[(size_t)(r + hg * 8) * QKDIM + qb * 16 + l16] = (_Float16)acc[r];
        }
    }
    __syncthreads();

    // ---- Phase 2: reload q as A-fragments (row-per-lane layout) ----
    v16h qf[8];
    {
        const _Float16* qrow = q_lds + wave * 16 * QKDIM + (size_t)l16 * QKDIM;
#pragma unroll
        for (int kc = 0; kc < 8; ++kc) {
            const int base = kc * 32 + hg * 16;
            const v8h lo = *(const v8h*)(qrow + base);
            const v8h hi = *(const v8h*)(qrow + base + 8);
            qf[kc] = __builtin_shufflevector(lo, hi, 0, 1, 2, 3, 4, 5, 6, 7,
                                             8, 9, 10, 11, 12, 13, 14, 15);
        }
    }

    // ---- Phase 3: scores = q @ k^T, fuse +gumbel and running argmax ----
    // gumbel is read exactly once kernel-wide -> non-temporal loads.
    float rmax[8];
    int   ridx[8];
#pragma unroll
    for (int r = 0; r < 8; ++r) { rmax[r] = -INFINITY; ridx[r] = 0; }

    const float* grow = gumbel + (size_t)row0 * NDIM;
#pragma unroll 4
    for (int nb = 0; nb < 32; ++nb) {
        v8f acc = {};
        const _Float16* kb = k_h + (size_t)(nb * 16 + l16) * QKDIM + hg * 16;
#pragma unroll
        for (int kc = 0; kc < 8; ++kc) {
            const v16h b = *(const v16h*)(kb + kc * 32);
            acc = __builtin_amdgcn_wmma_f32_16x16x32_f16(
                false, qf[kc], false, b, (short)0, acc, false, false);
        }
        const int n = nb * 16 + l16;
#pragma unroll
        for (int r = 0; r < 8; ++r) {
            const int m = r + hg * 8;
            const float g = __builtin_nontemporal_load(grow + (size_t)m * NDIM + n);
            const float logit = fmaf(acc[r], SCALING, g);
            if (logit > rmax[r]) { rmax[r] = logit; ridx[r] = n; }  // strict > : first max wins
        }
    }

    // ---- cross-lane argmax reduction within each 16-lane half (rows match) ----
#pragma unroll
    for (int msk = 1; msk <= 8; msk <<= 1) {
#pragma unroll
        for (int r = 0; r < 8; ++r) {
            const float om = __shfl_xor(rmax[r], msk, 32);
            const int   oi = __shfl_xor(ridx[r], msk, 32);
            if (om > rmax[r] || (om == rmax[r] && oi < ridx[r])) {
                rmax[r] = om; ridx[r] = oi;
            }
        }
    }

    // ---- fetch the other half-group's winners so every lane knows all 16 rows ----
    int oth[8];
#pragma unroll
    for (int r = 0; r < 8; ++r) oth[r] = __shfl_xor(ridx[r], 16, 32);

    // ---- epilogue: single-pass fused one-hot fill (NT stores, never re-read) ----
    // Row m winner: m<8 -> (hg ? oth[m] : ridx[m]); m>=8 -> (hg ? ridx[m-8] : oth[m-8]).
    {
        float* es = env_sel + (size_t)row0 * NDIM;
#pragma unroll
        for (int m = 0; m < 16; ++m) {
            const int id = (m < 8) ? (hg ? oth[m] : ridx[m])
                                   : (hg ? ridx[m - 8] : oth[m - 8]);
            float* rowp = es + (size_t)m * NDIM;
#pragma unroll
            for (int j = 0; j < 4; ++j) {
                const int c0 = (j * 32 + lane) * 4;       // column base, 0..508
                v4f val;
                val[0] = (id == c0 + 0) ? 1.0f : 0.0f;
                val[1] = (id == c0 + 1) ? 1.0f : 0.0f;
                val[2] = (id == c0 + 2) ? 1.0f : 0.0f;
                val[3] = (id == c0 + 3) ? 1.0f : 0.0f;
                __builtin_nontemporal_store(val, (v4f*)(rowp + c0));
            }
        }
    }

    // ---- env_att = sigmoid(v[argmax]), one lane per row ----
    if (l16 < 8) {
        const int mrow = hg * 8 + l16;                 // row within tile
        const int idx  = ridx[l16];
        const float vv = vvec[idx];
        env_att[row0 + mrow] = 1.0f / (1.0f + __expf(-vv));
    }
}

extern "C" void kernel_launch(void* const* d_in, const int* in_sizes, int n_in,
                              void* d_out, int out_size, void* d_ws, size_t ws_size,
                              hipStream_t stream) {
    const float* x      = (const float*)d_in[0];   // [32,2048,256]
    const float* env    = (const float*)d_in[1];   // [512,256]
    const float* Wq     = (const float*)d_in[2];   // [256,256]
    const float* Wk     = (const float*)d_in[3];   // [256,256]
    const float* Wv     = (const float*)d_in[4];   // [1,256]
    const float* gumbel = (const float*)d_in[5];   // [32,2048,512]

    // workspace: Wq_h (128KB) | k_h (256KB) | v (2KB)
    _Float16* Wq_h = (_Float16*)d_ws;
    _Float16* k_h  = (_Float16*)((char*)d_ws + (size_t)QKDIM * CDIM * 2);
    float*    vbuf = (float*)((char*)d_ws + (size_t)QKDIM * CDIM * 2 + (size_t)NDIM * DDIM * 2);

    float* env_att = (float*)d_out;        // [B,L,1]  = 65536 floats
    float* env_sel = env_att + BL;         // [B,L,N]  = 33554432 floats

    wq_to_f16<<<64, 256, 0, stream>>>(Wq, Wq_h);
    prep_kv<<<NDIM, 256, 0, stream>>>(env, Wk, Wv, k_h, vbuf);
    vq_cross_atten<<<BL / 128, 256, 0, stream>>>(x, Wq_h, k_h, vbuf, gumbel,
                                                 env_att, env_sel);
}
